// SegmentedAttention_54717883351441
// MI455X (gfx1250) — matrix operand order repaired and checked by reading, not compiled, not run on hardware
//
#include <hip/hip_runtime.h>
#include <cstddef>

// ---------------- problem constants ----------------
#define B_      2
#define SEQ_    4096
#define DIM_    1024
#define HEADS_  16
#define DH_     64
#define SEG_    512
#define NPM_    16
#define W_      8
#define KV_     (NPM_ + 2*SEG_)   // 1040
#define KV_TILES 33               // 33*32 = 1056 (pad last 16, masked)

typedef unsigned short us16;
typedef unsigned int   u32;
typedef __attribute__((ext_vector_type(4)))  u32     u32x4;
typedef __attribute__((ext_vector_type(4)))  float   f32x4;
typedef __attribute__((ext_vector_type(8)))  float   v8f;
typedef __attribute__((ext_vector_type(16))) __bf16  v16bf;

union FragH {            // one bf16 WMMA operand fragment: 8 VGPRs / lane
    v16bf v;
    u32x4 u[2];
    us16  s[16];
};

__device__ inline us16 f2bf(float f) {
    union { float f; u32 u; } c; c.f = f;
    u32 u = c.u + 0x7FFFu + ((c.u >> 16) & 1u);   // round-nearest-even
    return (us16)(u >> 16);
}
__device__ inline float bf2f(us16 h) {
    union { float f; u32 u; } c; c.u = ((u32)h) << 16;
    return c.f;
}
__device__ inline v8f zero8() { v8f z = {0.f,0.f,0.f,0.f,0.f,0.f,0.f,0.f}; return z; }

// Async Global->LDS 16-byte copy (CDNA5 GLOBAL_LOAD_ASYNC_TO_LDS_B128, ASYNCcnt).
// ldsp: generic pointer derived from __shared__ (low 32 bits = LDS byte offset).
__device__ inline void async_copy_b128(void* ldsp, const void* gptr) {
    unsigned lds_addr = (unsigned)(size_t)ldsp;
    unsigned long long g = (unsigned long long)gptr;
    asm volatile("global_load_async_to_lds_b128 %0, %1, off"
                 :: "v"(lds_addr), "v"(g) : "memory");
}
__device__ inline void async_wait0() {
    asm volatile("s_wait_asynccnt 0x0" ::: "memory");
}

// ---------------- elementwise fp32 -> bf16 cast ----------------
__global__ void cast_bf16_kernel(const float* __restrict__ in, us16* __restrict__ out, int n) {
    int i = blockIdx.x * blockDim.x + threadIdx.x;
    if (i < n) out[i] = f2bf(in[i]);
}

// ---------------- RMSNorm + cast (one row of 1024 per block) ----------------
__global__ __launch_bounds__(256)
void rmsnorm_kernel(const float* __restrict__ x, const float* __restrict__ g,
                    us16* __restrict__ out) {
    __shared__ float red[8];
    int row = blockIdx.x;
    int tid = threadIdx.x;
    const f32x4* xr = (const f32x4*)(x + (size_t)row * DIM_);
    f32x4 v = xr[tid];
    float ss = v.x*v.x + v.y*v.y + v.z*v.z + v.w*v.w;
    for (int off = 1; off < 32; off <<= 1) ss += __shfl_xor(ss, off, 32);
    if ((tid & 31) == 0) red[tid >> 5] = ss;
    __syncthreads();
    float tot = red[0]+red[1]+red[2]+red[3]+red[4]+red[5]+red[6]+red[7];
    float scale = rsqrtf(tot * (1.0f / DIM_) + 1.1920929e-07f);
    const f32x4* gr = (const f32x4*)g;
    f32x4 w = gr[tid];
    u32 p0 = (u32)f2bf(v.x*scale*w.x) | ((u32)f2bf(v.y*scale*w.y) << 16);
    u32 p1 = (u32)f2bf(v.z*scale*w.z) | ((u32)f2bf(v.w*scale*w.w) << 16);
    u32* dst = (u32*)(out + (size_t)row * DIM_);
    dst[tid*2+0] = p0;
    dst[tid*2+1] = p1;
}

// ---------------- generic bf16 GEMM: C[M,N] = A[M,K] * B[K,N] ----------------
// block tile 64(M) x 128(N), 8 waves -> 2x4 wave grid, 32x32 per wave (2x2 WMMA)
template <bool OUT_F32>
__global__ __launch_bounds__(256)
void gemm_bf16_kernel(const us16* __restrict__ A, const us16* __restrict__ Bm,
                      void* __restrict__ C, int M, int N, int K) {
    __shared__ __align__(16) us16 ldsA[64 * 40];    // 64 rows x 32 k, stride 40
    __shared__ __align__(16) us16 ldsBt[128 * 40];  // transposed: 128 n x 32 k
    int tid  = threadIdx.x;
    int wave = tid >> 5, lane = tid & 31, l16 = lane & 15, hi = lane >> 4;
    int wm = wave & 1, wn = wave >> 1;
    int m0 = blockIdx.y * 64, n0 = blockIdx.x * 128;

    v8f acc[2][2];
    for (int a = 0; a < 2; ++a) for (int b = 0; b < 2; ++b) acc[a][b] = zero8();

    for (int k0 = 0; k0 < K; k0 += 32) {
        // stage A tile (64x32): one async b128 Global->LDS per thread
        {
            int r  = tid >> 2;
            int cd = (tid & 3) * 8;
            async_copy_b128(ldsA + r * 40 + cd,
                            A + (size_t)(m0 + r) * K + k0 + cd);
        }
        // stage B tile transposed (32x128 -> [n][k]) through VGPRs
        for (int idx = tid; idx < 32 * 128; idx += 256) {
            int k = idx >> 7, n = idx & 127;
            ldsBt[n * 40 + k] = Bm[(size_t)(k0 + k) * N + n0 + n];
        }
        if (k0 + 32 < K)  // emits global_prefetch_b8
            __builtin_prefetch(A + (size_t)(m0 + (tid >> 2)) * K + k0 + 32, 0, 0);
        async_wait0();
        __syncthreads();

        FragH af[2], bf[2];
        for (int mi = 0; mi < 2; ++mi) {        // A 16x32 layout: lanes<16 K{0..7,16..23}
            int r  = wm * 32 + mi * 16 + l16;
            int d0 = hi * 8;
            af[mi].u[0] = *(const u32x4*)(ldsA + r * 40 + d0);
            af[mi].u[1] = *(const u32x4*)(ldsA + r * 40 + d0 + 16);
        }
        for (int ni = 0; ni < 2; ++ni) {        // B 32x16 layout: lanes<16 K{0..7,16..23}
            int n  = wn * 32 + ni * 16 + l16;
            int kk = hi * 8;
            bf[ni].u[0] = *(const u32x4*)(ldsBt + n * 40 + kk);
            bf[ni].u[1] = *(const u32x4*)(ldsBt + n * 40 + kk + 16);
        }
        for (int mi = 0; mi < 2; ++mi)
            for (int ni = 0; ni < 2; ++ni)
                acc[mi][ni] = __builtin_amdgcn_wmma_f32_16x16x32_bf16(
                    false, af[mi].v, false, bf[ni].v, (short)0, acc[mi][ni], false, false);
        __syncthreads();
    }
    // epilogue: C layout lane<16 -> M=g, lane>=16 -> M=8+g ; N = lane&15
    for (int mi = 0; mi < 2; ++mi)
        for (int ni = 0; ni < 2; ++ni) {
            int col = n0 + wn * 32 + ni * 16 + l16;
            for (int g = 0; g < 8; ++g) {
                int row = m0 + wm * 32 + mi * 16 + g + hi * 8;
                if constexpr (OUT_F32)
                    ((float*)C)[(size_t)row * N + col] = acc[mi][ni][g];
                else
                    ((us16*)C)[(size_t)row * N + col] = f2bf(acc[mi][ni][g]);
            }
        }
}

// ---------------- RoPE + scatter to (b,h,n,d) bf16 ----------------
__global__ __launch_bounds__(256)
void rope_kernel(const us16* __restrict__ qkv, us16* __restrict__ q,
                 us16* __restrict__ k, us16* __restrict__ v) {
    int bn = blockIdx.x;            // b*SEQ + n
    int b = bn >> 12, n = bn & (SEQ_ - 1);
    const us16* row = qkv + (size_t)bn * (3 * DIM_);
    for (int p = threadIdx.x; p < HEADS_ * DH_ / 2; p += 256) {
        int h = p >> 5, dp = p & 31;
        float inv = powf(10000.0f, -((float)(2 * dp) / 64.0f));
        float sn, cs;
        sincosf((float)n * inv, &sn, &cs);
        size_t dsti = (((size_t)(b * HEADS_ + h)) * SEQ_ + n) * DH_ + 2 * dp;
        int srci = h * DH_ + 2 * dp;
        { // q
            float x0 = bf2f(row[srci]), x1 = bf2f(row[srci + 1]);
            q[dsti]     = f2bf(x0 * cs - x1 * sn);
            q[dsti + 1] = f2bf(x1 * cs + x0 * sn);
        }
        { // k
            const us16* rk = row + DIM_;
            float x0 = bf2f(rk[srci]), x1 = bf2f(rk[srci + 1]);
            k[dsti]     = f2bf(x0 * cs - x1 * sn);
            k[dsti + 1] = f2bf(x1 * cs + x0 * sn);
        }
        { // v passthrough
            const us16* rv = row + 2 * DIM_;
            v[dsti]     = rv[srci];
            v[dsti + 1] = rv[srci + 1];
        }
    }
}

// ---------------- flash attention over one window/head ----------------
__device__ inline bool kv_visible(int i, int j, int w) {
    if (j < NPM_)        return true;
    if (j < NPM_ + SEG_) return (w > 0) && (i <= (j - NPM_));
    if (j < KV_)         return (j - NPM_ - SEG_) <= i;
    return false;  // pad
}
__device__ inline const us16* kv_row(int j, int w, const us16* pmrow, const us16* arr) {
    if (j < NPM_)        return pmrow + j * DH_;
    if (j < NPM_ + SEG_) return (w > 0) ? arr + ((size_t)((w - 1) * SEG_ + (j - NPM_))) * DH_
                                        : pmrow;              // masked dummy
    if (j < KV_)         return arr + ((size_t)(w * SEG_ + (j - NPM_ - SEG_))) * DH_;
    return pmrow;                                             // pad dummy
}

__global__ __launch_bounds__(128)
void attn_kernel(const us16* __restrict__ qg, const us16* __restrict__ kg,
                 const us16* __restrict__ vg, const us16* __restrict__ pmg,
                 us16* __restrict__ og) {
    // V staged TRANSPOSED: Vt[d][kv], stride 40 halves -> B-fragments are two
    // contiguous ds_load_b128 per lane (no scalar gather, no v_mov packing).
    __shared__ __align__(16) us16 ldsVt[4][64 * 40];
    __shared__ __align__(16) us16 ldsP[4][16 * 32];
    int tid  = threadIdx.x;
    int wave = tid >> 5, lane = tid & 31, l16 = lane & 15, hi = lane >> 4;

    int bx = blockIdx.x;
    int rb = bx & 7;
    int h  = (bx >> 3) & 15;
    int w  = (bx >> 7) & 7;
    int b  = bx >> 10;
    int rowBase = rb * 64 + wave * 16;   // query row tile within segment

    const us16* qb  = qg + (((size_t)(b * HEADS_ + h)) * SEQ_ + (size_t)w * SEG_ + rowBase) * DH_;
    const us16* kb  = kg + (((size_t)(b * HEADS_ + h)) * SEQ_) * DH_;
    const us16* vb  = vg + (((size_t)(b * HEADS_ + h)) * SEQ_) * DH_;
    const us16* pmk = pmg + (size_t)h * NPM_ * DH_;
    const us16* pmv = pmg + (size_t)(HEADS_ + h) * NPM_ * DH_;

    // Q fragments (held for the whole KV loop): 16 rows x 64 d  -> two 16x32 A frags
    FragH aq[2];
    for (int f = 0; f < 2; ++f) {
        int d0 = f * 32 + hi * 8;
        aq[f].u[0] = *(const u32x4*)(qb + l16 * DH_ + d0);
        aq[f].u[1] = *(const u32x4*)(qb + l16 * DH_ + d0 + 16);
    }

    float m[8], l[8];
    v8f accO[4];
    for (int g = 0; g < 8; ++g) { m[g] = -1e30f; l[g] = 0.0f; }
    for (int dt = 0; dt < 4; ++dt) accO[dt] = zero8();

    us16* myVt = ldsVt[wave];
    us16* myP  = ldsP[wave];

    for (int t = 0; t < KV_TILES; ++t) {
        int kv0 = t * 32;
        // stage V tile transposed: lane owns kv row (kv0+lane), writes column `lane`
        {
            const us16* src = kv_row(kv0 + lane, w, pmv, vb);
            #pragma unroll 8
            for (int d = 0; d < DH_; ++d)
                myVt[d * 40 + lane] = src[d];
        }
        // S = Q K^T for 16x32 tile, two 16-col WMMA n-tiles
        v8f s[2];
        for (int nt = 0; nt < 2; ++nt) {
            const us16* kr = kv_row(kv0 + nt * 16 + l16, w, pmk, kb);
            FragH bk0, bk1;   // B 32x16: lanes<16 d{0..7,16..23}
            bk0.u[0] = *(const u32x4*)(kr + hi * 8);
            bk0.u[1] = *(const u32x4*)(kr + hi * 8 + 16);
            bk1.u[0] = *(const u32x4*)(kr + 32 + hi * 8);
            bk1.u[1] = *(const u32x4*)(kr + 32 + hi * 8 + 16);
            v8f a = zero8();
            a = __builtin_amdgcn_wmma_f32_16x16x32_bf16(false, aq[0].v, false, bk0.v, (short)0, a, false, false);
            a = __builtin_amdgcn_wmma_f32_16x16x32_bf16(false, aq[1].v, false, bk1.v, (short)0, a, false, false);
            s[nt] = a;
        }
        // online softmax per row (rows split across lane halves)
        for (int g = 0; g < 8; ++g) {
            int i = rowBase + g + hi * 8;
            float sv0, sv1;
            {
                int j0 = kv0 + l16, j1 = kv0 + 16 + l16;
                sv0 = kv_visible(i, j0, w) ? s[0][g] * 0.125f : -1e30f;
                sv1 = kv_visible(i, j1, w) ? s[1][g] * 0.125f : -1e30f;
            }
            float loc = fmaxf(sv0, sv1);
            for (int off = 1; off < 16; off <<= 1) loc = fmaxf(loc, __shfl_xor(loc, off, 32));
            float mnew  = fmaxf(m[g], loc);
            float alpha = expf(m[g] - mnew);
            float p0 = expf(sv0 - mnew);
            float p1 = expf(sv1 - mnew);
            float rs = p0 + p1;
            for (int off = 1; off < 16; off <<= 1) rs += __shfl_xor(rs, off, 32);
            l[g] = l[g] * alpha + rs;
            m[g] = mnew;
            for (int dt = 0; dt < 4; ++dt) accO[dt][g] *= alpha;
            int r = g + hi * 8;
            myP[r * 32 + l16]      = f2bf(p0);
            myP[r * 32 + 16 + l16] = f2bf(p1);
        }
        __syncthreads();
        // P as A fragment (16x32)
        FragH aP;
        {
            int c0 = hi * 8;
            aP.u[0] = *(const u32x4*)(myP + l16 * 32 + c0);
            aP.u[1] = *(const u32x4*)(myP + l16 * 32 + c0 + 16);
        }
        // O += P * V : four 16-col d tiles; bV = two vector LDS loads from Vt
        for (int dt = 0; dt < 4; ++dt) {
            FragH bV;
            int d = dt * 16 + l16;
            bV.u[0] = *(const u32x4*)(myVt + d * 40 + hi * 8);
            bV.u[1] = *(const u32x4*)(myVt + d * 40 + hi * 8 + 16);
            accO[dt] = __builtin_amdgcn_wmma_f32_16x16x32_bf16(
                false, aP.v, false, bV.v, (short)0, accO[dt], false, false);
        }
        __syncthreads();
    }
    // epilogue: write O/l in (b, n, h*64+d) bf16 for the output projection
    for (int g = 0; g < 8; ++g) {
        int r = g + hi * 8;
        int n = w * SEG_ + rowBase + r;
        float inv = 1.0f / l[g];
        us16* orow = og + ((size_t)(b * SEQ_ + n)) * DIM_ + h * DH_;
        for (int dt = 0; dt < 4; ++dt)
            orow[dt * 16 + l16] = f2bf(accO[dt][g] * inv);
    }
}

// ---------------- host-side launch ----------------
extern "C" void kernel_launch(void* const* d_in, const int* in_sizes, int n_in,
                              void* d_out, int out_size, void* d_ws, size_t ws_size,
                              hipStream_t stream) {
    const float* seq    = (const float*)d_in[0];
    const float* norm_w = (const float*)d_in[1];
    const float* w_qkv  = (const float*)d_in[2];
    const float* w_out  = (const float*)d_in[3];
    const float* pmem   = (const float*)d_in[4];
    float* out = (float*)d_out;

    const size_t ROWS = (size_t)B_ * SEQ_;   // 8192
    us16* ws = (us16*)d_ws;
    size_t o = 0;
    us16* wqkv_bf = ws + o; o += (size_t)DIM_ * 3 * DIM_;        // 1024*3072
    us16* wout_bf = ws + o; o += (size_t)DIM_ * DIM_;            // 1024*1024
    us16* pm_bf   = ws + o; o += (size_t)2 * HEADS_ * NPM_ * DH_;
    us16* xn_bf   = ws + o; o += ROWS * DIM_;
    us16* qkv_bf  = ws + o; o += ROWS * 3 * DIM_;
    us16* q_bf    = ws + o; o += ROWS * DIM_;
    us16* k_bf    = ws + o; o += ROWS * DIM_;
    us16* v_bf    = ws + o; o += ROWS * DIM_;
    us16* o_bf    = ws + o; o += ROWS * DIM_;

    // weight / persistent-memory casts
    {
        int n = DIM_ * 3 * DIM_;
        cast_bf16_kernel<<<(n + 255) / 256, 256, 0, stream>>>(w_qkv, wqkv_bf, n);
    }
    {
        int n = DIM_ * DIM_;
        cast_bf16_kernel<<<(n + 255) / 256, 256, 0, stream>>>(w_out, wout_bf, n);
    }
    {
        int n = 2 * HEADS_ * NPM_ * DH_;
        cast_bf16_kernel<<<(n + 255) / 256, 256, 0, stream>>>(pmem, pm_bf, n);
    }
    // RMSNorm
    rmsnorm_kernel<<<(int)ROWS, 256, 0, stream>>>(seq, norm_w, xn_bf);
    // QKV projection (bf16 out)
    gemm_bf16_kernel<false><<<dim3(3 * DIM_ / 128, (int)ROWS / 64), 256, 0, stream>>>(
        xn_bf, wqkv_bf, (void*)qkv_bf, (int)ROWS, 3 * DIM_, DIM_);
    // RoPE + scatter
    rope_kernel<<<(int)ROWS, 256, 0, stream>>>(qkv_bf, q_bf, k_bf, v_bf);
    // flash attention
    attn_kernel<<<B_ * W_ * HEADS_ * (SEG_ / 64), 128, 0, stream>>>(
        q_bf, k_bf, v_bf, pm_bf, o_bf);
    // output projection (f32 out)
    gemm_bf16_kernel<true><<<dim3(DIM_ / 128, (int)ROWS / 64), 256, 0, stream>>>(
        o_bf, wout_bf, (void*)out, (int)ROWS, DIM_, DIM_);
}
